// CPUEfficientMoE_31920196944052
// MI455X (gfx1250) — compile-verified, matrix operand order checked by reading, not run on hardware
//
#include <hip/hip_runtime.h>
#include <hip/hip_bf16.h>
#include <stdint.h>

// ---------------------------------------------------------------------------
// MoE (top-2 of 8 experts), grouped-GEMM formulation for MI455X (gfx1250).
// bf16 WMMA 16x16x32, f32 accumulate. Wave32.
// TM=64 rows per tile: each B fragment feeds 4 WMMAs (32 FLOP per L2 byte ->
// firmly WMMA-bound). X tile and H tile share one 128KB dynamic-LDS buffer
// (X is dead after GEMM1's K-loop; ISA allows 320KB LDS per workgroup).
// ---------------------------------------------------------------------------

#define NTOK   8192
#define CDIM   1024
#define FDIM   1024
#define NEXP   8
#define TM     64                       // tile rows (slots) per workgroup
#define MTILES (TM / 16)                // 4 m-tiles per wave
#define SLOT_CAP (2*NTOK + NEXP*TM)     // padded segments fit
#define MAX_TILES ((2*NTOK)/TM + NEXP)  // 264 fixed upper bound
#define LDS_BYTES (TM * CDIM * 2)       // 128 KB

typedef __bf16 bf16_t;
typedef bf16_t v16bf __attribute__((ext_vector_type(16)));
typedef float  v8f   __attribute__((ext_vector_type(8)));

union FragU { uint4 q[2]; v16bf v; };

static __device__ inline unsigned short f2bf(float f) {
    unsigned u = __float_as_uint(f);
    unsigned r = ((u >> 16) & 1u) + 0x7FFFu;       // round-to-nearest-even
    return (unsigned short)((u + r) >> 16);
}

// Load a 16x32 bf16 WMMA fragment (A or B) where the 16-dim is "col" and the
// 32-dim (K) is contiguous in memory with row stride `stride` (elements).
// Lane layout per CDNA5 ISA 7.12.2: lanes 0-15 hold K {0..7,16..23},
// lanes 16-31 hold K {8..15,24..31}, for row/col = lane%16.
static __device__ inline v16bf load_frag(const unsigned short* base, int col,
                                         int stride, int k0, int lane) {
    int half = (lane & 16) ? 8 : 0;
    const unsigned short* p = base + (size_t)(col + (lane & 15)) * stride + k0 + half;
    FragU f;
    f.q[0] = *(const uint4*)(p);
    f.q[1] = *(const uint4*)(p + 16);
    return f.v;
}

// --------------------------- conversion kernels ----------------------------

__global__ __launch_bounds__(256) void convx_kernel(const float* __restrict__ in,
                                                    unsigned short* __restrict__ outp,
                                                    int n4) {
    int i = blockIdx.x * 256 + threadIdx.x;
    if (i >= n4) return;
    float4 v = ((const float4*)in)[i];
    union { unsigned short s[4]; uint2 u; } r;
    r.s[0] = f2bf(v.x); r.s[1] = f2bf(v.y); r.s[2] = f2bf(v.z); r.s[3] = f2bf(v.w);
    ((uint2*)outp)[i] = r.u;
}

// out[mat][c][r] = bf16(in[mat][r][c]); R == Ccols == 1024 here.
__global__ __launch_bounds__(256) void transpose_bf16_kernel(const float* __restrict__ in,
                                                             unsigned short* __restrict__ outp,
                                                             int R, int Ccols) {
    __shared__ float tile[32][33];
    int mat = blockIdx.z;
    const float* src = in + (size_t)mat * R * Ccols;
    unsigned short* dst = outp + (size_t)mat * R * Ccols;
    int c0 = blockIdx.x * 32, r0 = blockIdx.y * 32;
    for (int i = threadIdx.y; i < 32; i += 8)
        tile[i][threadIdx.x] = src[(size_t)(r0 + i) * Ccols + c0 + threadIdx.x];
    __syncthreads();
    for (int i = threadIdx.y; i < 32; i += 8)
        dst[(size_t)(c0 + i) * R + r0 + threadIdx.x] = f2bf(tile[threadIdx.x][i]);
}

// ------------------------------ router -------------------------------------

__global__ __launch_bounds__(256) void router_kernel(const float* __restrict__ x,
                                                     const float* __restrict__ rw,
                                                     int2* __restrict__ route_idx,
                                                     float2* __restrict__ route_w,
                                                     int* __restrict__ counts) {
    __shared__ float rws[NEXP * CDIM];
    for (int i = threadIdx.x; i < NEXP * CDIM; i += 256) rws[i] = rw[i];
    __syncthreads();

    int wave = threadIdx.x >> 5, lane = threadIdx.x & 31;
    int tok = blockIdx.x * 8 + wave;
    if (tok >= NTOK) return;

    float acc[NEXP] = {};
    const float* xr = x + (size_t)tok * CDIM;
    for (int c = lane; c < CDIM; c += 32) {
        float xv = xr[c];
        #pragma unroll
        for (int e = 0; e < NEXP; e++) acc[e] += xv * rws[e * CDIM + c];
    }
    #pragma unroll
    for (int e = 0; e < NEXP; e++)
        #pragma unroll
        for (int off = 16; off >= 1; off >>= 1)
            acc[e] += __shfl_xor(acc[e], off, 32);

    if (lane == 0) {
        float m = acc[0];
        #pragma unroll
        for (int e = 1; e < NEXP; e++) m = fmaxf(m, acc[e]);
        float p[NEXP], s = 0.f;
        #pragma unroll
        for (int e = 0; e < NEXP; e++) { p[e] = __expf(acc[e] - m); s += p[e]; }
        float inv = 1.0f / s;
        int i0 = 0;
        for (int e = 1; e < NEXP; e++) if (p[e] > p[i0]) i0 = e;   // low-index tie-break
        int i1 = -1;
        for (int e = 0; e < NEXP; e++) {
            if (e == i0) continue;
            if (i1 < 0 || p[e] > p[i1]) i1 = e;
        }
        route_idx[tok] = make_int2(i0, i1);
        route_w[tok]   = make_float2(p[i0] * inv, p[i1] * inv);
        atomicAdd(&counts[i0], 1);
        atomicAdd(&counts[i1], 1);
    }
}

// offsets padded to multiples of TM so each tile lies inside one expert segment
__global__ void offs_kernel(const int* __restrict__ counts, int* __restrict__ offs) {
    if (threadIdx.x == 0 && blockIdx.x == 0) {
        int o = 0; offs[0] = 0;
        for (int e = 0; e < NEXP; e++) {
            o += ((counts[e] + TM - 1) / TM) * TM;
            offs[e + 1] = o;
        }
    }
}

__global__ __launch_bounds__(256) void scatter_kernel(const int2* __restrict__ ridx,
                                                      const float2* __restrict__ rwt,
                                                      const int* __restrict__ offs,
                                                      int* __restrict__ fill,
                                                      int* __restrict__ slot_token,
                                                      float* __restrict__ slot_gate) {
    int t = blockIdx.x * 256 + threadIdx.x;
    if (t >= NTOK) return;
    int2 id = ridx[t];
    float2 wt = rwt[t];
    int p0 = atomicAdd(&fill[id.x], 1);
    int s0 = offs[id.x] + p0;
    slot_token[s0] = t; slot_gate[s0] = wt.x;
    int p1 = atomicAdd(&fill[id.y], 1);
    int s1 = offs[id.y] + p1;
    slot_token[s1] = t; slot_gate[s1] = wt.y;
}

// --------------------------- grouped FFN GEMM ------------------------------
// Per block: one 64-slot tile of a single expert.
//   GEMM1: H = relu(X[64,1024] @ W1[e])   (bf16 WMMA, f32 acc)
//   GEMM2: Y = H @ W2[e]; out[token] += gate * Y  (atomic f32, commutative)
// Each wave: 128 output cols (8 n-tiles) x 4 m-tiles; B fragment reused 4x.
__global__ __launch_bounds__(256) void ffn_kernel(const unsigned short* __restrict__ xb,   // [N][C]
                                                  const unsigned short* __restrict__ w1t,  // [E][F][C]
                                                  const unsigned short* __restrict__ w2t,  // [E][C][F]
                                                  const int* __restrict__ slot_token,
                                                  const float* __restrict__ slot_gate,
                                                  const int* __restrict__ offs,
                                                  float* __restrict__ out) {
    extern __shared__ unsigned short Buf[];    // 128 KB: X tile, then H tile (overlaid)

    int tile0 = blockIdx.x * TM;
    if (tile0 >= offs[NEXP]) return;
    int e = 0;
    while (tile0 >= offs[e + 1]) e++;

    int tid  = threadIdx.x;
    int lane = tid & 31;
    int wave = tid >> 5;

    // Stage X tile (uint4 = 8 bf16 per transaction); invalid slots -> zeros.
    for (int i = tid; i < TM * (CDIM / 8); i += 256) {
        int row = i / (CDIM / 8);
        int kc  = (i % (CDIM / 8)) * 8;
        int tok = slot_token[tile0 + row];
        uint4 val = make_uint4(0u, 0u, 0u, 0u);
        if (tok >= 0) val = *(const uint4*)(xb + (size_t)tok * CDIM + kc);
        *(uint4*)(Buf + row * CDIM + kc) = val;
    }
    __syncthreads();

    const int nbase = wave * 128;                 // each wave: 128 output cols
    const int mhalf = (lane & 16) ? 8 : 0;
    const int ncol0 = lane & 15;

    // ---------------- GEMM1: hidden ----------------
    const unsigned short* w1e = w1t + (size_t)e * FDIM * CDIM;
    v8f acc[MTILES][8];
    #pragma unroll
    for (int mt = 0; mt < MTILES; mt++)
        #pragma unroll
        for (int nt = 0; nt < 8; nt++) acc[mt][nt] = (v8f){0,0,0,0,0,0,0,0};

    for (int k0 = 0; k0 < CDIM; k0 += 32) {
        v16bf a[MTILES];
        #pragma unroll
        for (int mt = 0; mt < MTILES; mt++)
            a[mt] = load_frag(Buf, mt * 16, CDIM, k0, lane);
        #pragma unroll
        for (int nt = 0; nt < 8; nt++) {
            v16bf b = load_frag(w1e, nbase + nt * 16, CDIM, k0, lane);
            #pragma unroll
            for (int mt = 0; mt < MTILES; mt++)
                acc[mt][nt] = __builtin_amdgcn_wmma_f32_16x16x32_bf16(
                    false, a[mt], false, b, (short)0, acc[mt][nt], false, false);
        }
    }
    __syncthreads();   // all waves done reading X -> safe to overwrite with H

    // relu -> bf16 -> LDS (overlaid on X buffer)
    #pragma unroll
    for (int mt = 0; mt < MTILES; mt++)
        #pragma unroll
        for (int nt = 0; nt < 8; nt++) {
            int col = nbase + nt * 16 + ncol0;
            #pragma unroll
            for (int v = 0; v < 8; v++) {
                float h = acc[mt][nt][v];
                h = h > 0.f ? h : 0.f;
                Buf[(mt * 16 + mhalf + v) * FDIM + col] = f2bf(h);
            }
        }
    __syncthreads();

    // ---------------- GEMM2: output ----------------
    const unsigned short* w2e = w2t + (size_t)e * CDIM * FDIM;
    #pragma unroll
    for (int mt = 0; mt < MTILES; mt++)
        #pragma unroll
        for (int nt = 0; nt < 8; nt++) acc[mt][nt] = (v8f){0,0,0,0,0,0,0,0};

    for (int k0 = 0; k0 < FDIM; k0 += 32) {
        v16bf a[MTILES];
        #pragma unroll
        for (int mt = 0; mt < MTILES; mt++)
            a[mt] = load_frag(Buf, mt * 16, FDIM, k0, lane);
        #pragma unroll
        for (int nt = 0; nt < 8; nt++) {
            v16bf b = load_frag(w2e, nbase + nt * 16, FDIM, k0, lane);
            #pragma unroll
            for (int mt = 0; mt < MTILES; mt++)
                acc[mt][nt] = __builtin_amdgcn_wmma_f32_16x16x32_bf16(
                    false, a[mt], false, b, (short)0, acc[mt][nt], false, false);
        }
    }

    // gate-scaled scatter-accumulate (2 commutative f32 adds per element)
    #pragma unroll
    for (int mt = 0; mt < MTILES; mt++)
        #pragma unroll
        for (int v = 0; v < 8; v++) {
            int s   = tile0 + mt * 16 + mhalf + v;
            int tok = slot_token[s];
            if (tok < 0) continue;
            float g = slot_gate[s];
            #pragma unroll
            for (int nt = 0; nt < 8; nt++) {
                int col = nbase + nt * 16 + ncol0;
                atomicAdd(out + (size_t)tok * CDIM + col, g * acc[mt][nt][v]);
            }
        }
}

// ------------------------------ launcher -----------------------------------

extern "C" void kernel_launch(void* const* d_in, const int* in_sizes, int n_in,
                              void* d_out, int out_size, void* d_ws, size_t ws_size,
                              hipStream_t stream) {
    (void)in_sizes; (void)n_in; (void)out_size; (void)ws_size;
    const float* x        = (const float*)d_in[0];
    const float* router_w = (const float*)d_in[1];
    const float* w1       = (const float*)d_in[2];
    const float* w2       = (const float*)d_in[3];
    float* out = (float*)d_out;

    // Workspace carving (256B aligned)
    char* ws = (char*)d_ws;
    size_t off = 0;
    auto carve = [&](size_t bytes) -> void* {
        off = (off + 255) & ~(size_t)255;
        void* p = ws + off;
        off += bytes;
        return p;
    };
    unsigned short* xb   = (unsigned short*)carve((size_t)NTOK * CDIM * 2);
    unsigned short* w1t  = (unsigned short*)carve((size_t)NEXP * FDIM * CDIM * 2);
    unsigned short* w2t  = (unsigned short*)carve((size_t)NEXP * CDIM * FDIM * 2);
    int2*   route_idx    = (int2*)  carve((size_t)NTOK * sizeof(int2));
    float2* route_w      = (float2*)carve((size_t)NTOK * sizeof(float2));
    int*    counts       = (int*)   carve(NEXP * sizeof(int));
    int*    offs         = (int*)   carve((NEXP + 1) * sizeof(int));
    int*    fill         = (int*)   carve(NEXP * sizeof(int));
    int*    slot_token   = (int*)   carve((size_t)SLOT_CAP * sizeof(int));
    float*  slot_gate    = (float*) carve((size_t)SLOT_CAP * sizeof(float));

    hipMemsetAsync(counts, 0, NEXP * sizeof(int), stream);
    hipMemsetAsync(fill,   0, NEXP * sizeof(int), stream);
    hipMemsetAsync(slot_token, 0xFF, (size_t)SLOT_CAP * sizeof(int), stream);  // -1
    hipMemsetAsync(out, 0, (size_t)NTOK * CDIM * sizeof(float), stream);

    // 1) precision conversion / weight layout
    {
        int n4 = (NTOK * CDIM) / 4;
        convx_kernel<<<(n4 + 255) / 256, 256, 0, stream>>>(x, xb, n4);
        dim3 tb(32, 8, 1);
        dim3 tg(CDIM / 32, FDIM / 32, NEXP);
        transpose_bf16_kernel<<<tg, tb, 0, stream>>>(w1, w1t, CDIM, FDIM); // -> [E][F][C]
        transpose_bf16_kernel<<<tg, tb, 0, stream>>>(w2, w2t, FDIM, CDIM); // -> [E][C][F]
    }

    // 2) router (softmax + top-2) and expert counts
    router_kernel<<<NTOK / 8, 256, 0, stream>>>(x, router_w, route_idx, route_w, counts);

    // 3) padded per-expert offsets, 4) scatter tokens to slots
    offs_kernel<<<1, 32, 0, stream>>>(counts, offs);
    scatter_kernel<<<(NTOK + 255) / 256, 256, 0, stream>>>(route_idx, route_w, offs,
                                                           fill, slot_token, slot_gate);

    // 5) grouped FFN GEMM (fixed tile-count upper bound; empty tiles early-out)
    ffn_kernel<<<MAX_TILES, 256, LDS_BYTES, stream>>>(xb, w1t, w2t, slot_token,
                                                      slot_gate, offs, out);
}
